// TransformerBlock_32401233281677
// MI455X (gfx1250) — matrix operand order repaired and checked by reading, not compiled, not run on hardware
//
#include <hip/hip_runtime.h>
#include <hip/hip_bf16.h>
#include <math.h>

// ---------------- types & constants ----------------
typedef __attribute__((ext_vector_type(16))) __bf16 v16bf;
typedef __attribute__((ext_vector_type(2)))  __bf16 v2bf;
typedef __attribute__((ext_vector_type(8)))  float  v8f;

#define S_LEN 2048
#define C_DIM 1024
#define M_DIM 4096
#define NHEAD 16
#define HD    64

#define TM 128
#define TN 128
#define TK 64
#define APAD 72   // padded bf16 row stride for A tile in LDS (64 + 8)
#define BPAD 72   // padded bf16 row stride for B (transposed) tile in LDS

union bfvec16 { v16bf v; __bf16 e[16]; };

static __device__ inline float gelu_exact(float x) {
  return 0.5f * x * (1.0f + erff(x * 0.70710678118654752f));
}

// ---------------- LayerNorm: one block per row ----------------
__global__ void __launch_bounds__(256) ln_kernel(const float* __restrict__ x,
                                                 const float* __restrict__ g,
                                                 const float* __restrict__ b,
                                                 float* __restrict__ y) {
  __shared__ float red[256];
  int row = blockIdx.x, t = threadIdx.x;
  const float* xr = x + (size_t)row * C_DIM;
  float v0 = xr[t], v1 = xr[t + 256], v2 = xr[t + 512], v3 = xr[t + 768];
  red[t] = v0 + v1 + v2 + v3;
  __syncthreads();
  for (int o = 128; o > 0; o >>= 1) { if (t < o) red[t] += red[t + o]; __syncthreads(); }
  float mean = red[0] * (1.0f / C_DIM);
  __syncthreads();
  float d0 = v0 - mean, d1 = v1 - mean, d2 = v2 - mean, d3 = v3 - mean;
  red[t] = d0 * d0 + d1 * d1 + d2 * d2 + d3 * d3;
  __syncthreads();
  for (int o = 128; o > 0; o >>= 1) { if (t < o) red[t] += red[t + o]; __syncthreads(); }
  float rstd = rsqrtf(red[0] * (1.0f / C_DIM) + 1e-6f);
  float* yr = y + (size_t)row * C_DIM;
  yr[t]       = d0 * rstd * g[t]       + b[t];
  yr[t + 256] = d1 * rstd * g[t + 256] + b[t + 256];
  yr[t + 512] = d2 * rstd * g[t + 512] + b[t + 512];
  yr[t + 768] = d3 * rstd * g[t + 768] + b[t + 768];
}

// ---------------- GEMM: Y = epilogue(X[MxK] * W[KxN] + bias) ----------------
// MODE 0: bias only      MODE 1: bias + exact GELU
// MODE 2: bias + residual R[MxN] (f32 out)
// MODE 3: (bias'ed) * scale, BF16 head-major   [H][M][64]   (for Q, K)
// MODE 4: (bias'ed),         BF16 head-major-T [H][64][M]   (for V)
template<int MODE>
__global__ void __launch_bounds__(256) gemm_kernel(
    const float* __restrict__ X, const float* __restrict__ W,
    const float* __restrict__ bias, const float* __restrict__ R,
    float* __restrict__ Y, int M, int N, int K, float scale) {
  __shared__ __bf16 As[TM * APAD];   // [m][k] row-major
  __shared__ __bf16 Bs[TN * BPAD];   // [n][k] (transposed) so B-frag loads are contiguous
  int t = threadIdx.x;
  int m0 = blockIdx.y * TM, n0 = blockIdx.x * TN;
  int w = t >> 5, lane = t & 31, l16 = lane & 15, hi = lane >> 4;
  int wr = (w & 3) * 32;   // wave row offset within tile (4 row-waves)
  int wc = (w >> 2) * 64;  // wave col offset within tile (2 col-waves)

  const v8f vzero = {0.f, 0.f, 0.f, 0.f, 0.f, 0.f, 0.f, 0.f};
  v8f acc[2][4];
  #pragma unroll
  for (int i = 0; i < 2; ++i)
    #pragma unroll
    for (int j = 0; j < 4; ++j) acc[i][j] = vzero;

  // A staging: 128 rows x 64 k, 32 elems/thread (half-row chunk)
  int arow = t >> 1, akoff = (t & 1) * 32;
  // B staging: 64 k x 128 n; each thread owns 2 consecutive k x 16 n
  int bk = (t & 31) * 2, bn = (t >> 5) * 16;

  for (int k0 = 0; k0 < K; k0 += TK) {
    const float* ag = X + (size_t)(m0 + arow) * K + k0 + akoff;
    #pragma unroll
    for (int j = 0; j < 32; j += 2) {
      v2bf p; p[0] = (__bf16)ag[j]; p[1] = (__bf16)ag[j + 1];   // packed cvt
      *(v2bf*)&As[arow * APAD + akoff + j] = p;                 // 32-bit DS store
    }
    const float* bg0 = W + (size_t)(k0 + bk) * N + n0 + bn;
    const float* bg1 = bg0 + N;
    #pragma unroll
    for (int j = 0; j < 16; ++j) {
      v2bf p; p[0] = (__bf16)bg0[j]; p[1] = (__bf16)bg1[j];
      *(v2bf*)&Bs[(bn + j) * BPAD + bk] = p;                    // transposed [n][k]
    }
    if (k0 + TK < K) {                       // gfx1250 global_prefetch_b8 path
      __builtin_prefetch(ag + TK, 0, 1);
      __builtin_prefetch(bg0 + (size_t)TK * N, 0, 1);
    }
    __syncthreads();

    #pragma unroll
    for (int ks = 0; ks < 2; ++ks) {         // two 32-deep WMMA slices per panel
      v16bf afrag[2];
      #pragma unroll
      for (int rt = 0; rt < 2; ++rt) {
        // A frag, contiguous k order (same as B): lanes 0-15 rows M=0..15 K=0..15; lanes 16-31 K=16..31
        const __bf16* ar = &As[(wr + rt * 16 + l16) * APAD + ks * 32 + (hi ? 16 : 0)];
        bfvec16 u;
        __builtin_memcpy(&u.e[0], ar, 16);
        __builtin_memcpy(&u.e[8], ar + 8, 16);
        afrag[rt] = u.v;
      }
      #pragma unroll
      for (int ct = 0; ct < 4; ++ct) {
        // B frag, contiguous k order (same as A): lane n = l16, lanes 0-15 K=0..15, lanes 16-31 K=16..31
        const __bf16* br = &Bs[(wc + ct * 16 + l16) * BPAD + ks * 32 + (hi ? 16 : 0)];
        bfvec16 ub;
        __builtin_memcpy(ub.e, br, 32);
        acc[0][ct] = __builtin_amdgcn_wmma_f32_16x16x32_bf16(
            false, afrag[0], false, ub.v, (short)0, acc[0][ct], false, false);
        acc[1][ct] = __builtin_amdgcn_wmma_f32_16x16x32_bf16(
            false, afrag[1], false, ub.v, (short)0, acc[1][ct], false, false);
      }
    }
    __syncthreads();
  }

  // epilogue: C layout — vgpr r holds (m = r + 8*hi, n = l16)
  #pragma unroll
  for (int rt = 0; rt < 2; ++rt) {
    #pragma unroll
    for (int ct = 0; ct < 4; ++ct) {
      int n = n0 + wc + ct * 16 + l16;
      int mb = m0 + wr + rt * 16 + hi * 8;
      float bv = bias[n];
      if (MODE == 4) {
        // V^T: [H][64][M]; per-lane rows m are contiguous -> paired bf16 stores
        __bf16* yt = (__bf16*)Y + ((size_t)(n >> 6) * HD + (n & 63)) * M + mb;
        #pragma unroll
        for (int r = 0; r < 8; r += 2) {
          v2bf p; p[0] = (__bf16)(acc[rt][ct][r] + bv);
          p[1] = (__bf16)(acc[rt][ct][r + 1] + bv);
          *(v2bf*)&yt[r] = p;
        }
      } else {
        #pragma unroll
        for (int r = 0; r < 8; ++r) {
          int m = mb + r;
          float val = acc[rt][ct][r] + bv;
          if (MODE == 1) val = gelu_exact(val);
          if (MODE == 2) val += R[(size_t)m * N + n];
          if (MODE == 3) {
            val *= scale;
            ((__bf16*)Y)[((size_t)(n >> 6) * M + m) * HD + (n & 63)] = (__bf16)val;
          } else {
            Y[(size_t)m * N + n] = val;
          }
        }
      }
    }
  }
}

// ---------------- Flash attention ----------------
// q,k BF16 [H][S][64]; v BF16 transposed [H][64][S]; ctx f32 (S, C)
template<int CAUSAL>
__global__ void __launch_bounds__(256) attn_kernel(
    const __bf16* __restrict__ q, const __bf16* __restrict__ k,
    const __bf16* __restrict__ v, float* __restrict__ ctx) {
  __shared__ __bf16 Pl[8][16 * 34];   // per-wave private P-transpose bounce buffer
  int tid = threadIdx.x;
  int w = tid >> 5, lane = tid & 31, l16 = lane & 15, hi = lane >> 4;
  int h = blockIdx.y;
  int qbase = blockIdx.x * 128;
  int qrow = qbase + w * 16;          // each wave owns 16 query rows
  const __bf16* qh = q + (size_t)h * S_LEN * HD;
  const __bf16* kh = k + (size_t)h * S_LEN * HD;
  const __bf16* vh = v + (size_t)h * HD * S_LEN;   // transposed

  // Q A-frags, two hd-slices of 32
  v16bf qa[2];
  #pragma unroll
  for (int s = 0; s < 2; ++s) {
    const __bf16* qr = qh + (size_t)(qrow + l16) * HD + s * 32 + (hi ? 16 : 0);
    bfvec16 u;
    __builtin_memcpy(&u.e[0], qr, 16);
    __builtin_memcpy(&u.e[8], qr + 8, 16);
    qa[s] = u.v;
  }

  const v8f vzero = {0.f, 0.f, 0.f, 0.f, 0.f, 0.f, 0.f, 0.f};
  v8f o[4];
  #pragma unroll
  for (int tt = 0; tt < 4; ++tt) o[tt] = vzero;
  float mrow[8], lrow[8];
  #pragma unroll
  for (int r = 0; r < 8; ++r) { mrow[r] = -1e30f; lrow[r] = 0.f; }

  int kend = CAUSAL ? (qbase + 128) : S_LEN;
  for (int kb0 = 0; kb0 < kend; kb0 += 32) {
    // ---- S = Q * K^T  (16x32 scores = two 16x16 C tiles) ----
    v8f st[2]; st[0] = vzero; st[1] = vzero;
    #pragma unroll
    for (int tt = 0; tt < 2; ++tt) {
      #pragma unroll
      for (int s = 0; s < 2; ++s) {
        const __bf16* kr = kh + (size_t)(kb0 + tt * 16 + l16) * HD + s * 32 + (hi ? 16 : 0);
        bfvec16 ub;
        __builtin_memcpy(ub.e, kr, 32);         // contiguous 32B
        st[tt] = __builtin_amdgcn_wmma_f32_16x16x32_bf16(
            false, qa[s], false, ub.v, (short)0, st[tt], false, false);
      }
    }
    // ---- causal mask: only diagonal-touching tiles (wave-uniform branch) ----
    if (CAUSAL) {
      if (kb0 + 31 > qrow) {                    // wave-uniform: one s_cbranch per tile
        #pragma unroll
        for (int r = 0; r < 8; ++r) {
          int m = qrow + r + hi * 8;
          if (kb0 + l16 > m)      st[0][r] = -1e30f;
          if (kb0 + 16 + l16 > m) st[1][r] = -1e30f;
        }
      }
    }
    // ---- online softmax ----
    float rmax[8];
    #pragma unroll
    for (int r = 0; r < 8; ++r) rmax[r] = fmaxf(st[0][r], st[1][r]);
    #pragma unroll
    for (int off = 1; off < 16; off <<= 1)
      #pragma unroll
      for (int r = 0; r < 8; ++r)
        rmax[r] = fmaxf(rmax[r], __shfl_xor(rmax[r], off, 32));  // stays in 16-lane row group
    float corr[8], rsum[8];
    #pragma unroll
    for (int r = 0; r < 8; ++r) {
      float mnew = fmaxf(mrow[r], rmax[r]);
      corr[r] = __expf(mrow[r] - mnew);
      mrow[r] = mnew;
      float p0 = __expf(st[0][r] - mnew), p1 = __expf(st[1][r] - mnew);
      st[0][r] = p0; st[1][r] = p1;
      rsum[r] = p0 + p1;
    }
    #pragma unroll
    for (int off = 1; off < 16; off <<= 1)
      #pragma unroll
      for (int r = 0; r < 8; ++r)
        rsum[r] += __shfl_xor(rsum[r], off, 32);
    #pragma unroll
    for (int r = 0; r < 8; ++r) lrow[r] = lrow[r] * corr[r] + rsum[r];
    #pragma unroll
    for (int tt = 0; tt < 4; ++tt)
      #pragma unroll
      for (int r = 0; r < 8; ++r) o[tt][r] *= corr[r];

    // ---- P (C layout) -> LDS -> A layout; wave-private, DS ops in-order per wave ----
    #pragma unroll
    for (int r = 0; r < 8; ++r) {
      Pl[w][(r + hi * 8) * 34 + l16]      = (__bf16)st[0][r];
      Pl[w][(r + hi * 8) * 34 + 16 + l16] = (__bf16)st[1][r];
    }
    const __bf16* pr = &Pl[w][l16 * 34 + (hi ? 16 : 0)];
    bfvec16 up;
    __builtin_memcpy(&up.e[0], pr, 16);
    __builtin_memcpy(&up.e[8], pr + 8, 16);
    v16bf pa = up.v;

    // ---- O += P * V   (V^T gives contiguous B-fragments) ----
    #pragma unroll
    for (int tt = 0; tt < 4; ++tt) {
      const __bf16* vr = vh + (size_t)(tt * 16 + l16) * S_LEN + kb0 + (hi ? 16 : 0);
      bfvec16 uv;
      __builtin_memcpy(uv.e, vr, 32);           // contiguous 32B
      o[tt] = __builtin_amdgcn_wmma_f32_16x16x32_bf16(
          false, pa, false, uv.v, (short)0, o[tt], false, false);
    }
  }

  // ---- normalize, merge heads into (S, C) ----
  float inv[8];
  #pragma unroll
  for (int r = 0; r < 8; ++r) inv[r] = 1.0f / lrow[r];
  #pragma unroll
  for (int tt = 0; tt < 4; ++tt)
    #pragma unroll
    for (int r = 0; r < 8; ++r) {
      int m = qrow + r + hi * 8;
      int d = tt * 16 + l16;
      ctx[(size_t)m * C_DIM + h * HD + d] = o[tt][r] * inv[r];
    }
}

// ---------------- orchestration ----------------
extern "C" void kernel_launch(void* const* d_in, const int* in_sizes, int n_in,
                              void* d_out, int out_size, void* d_ws, size_t ws_size,
                              hipStream_t stream) {
  const float* feat   = (const float*)d_in[0];
  const float* enc    = (const float*)d_in[1];
  const float* ln1a_g = (const float*)d_in[2],  *ln1a_b = (const float*)d_in[3];
  const float* ln1b_g = (const float*)d_in[4],  *ln1b_b = (const float*)d_in[5];
  const float* ln2a_g = (const float*)d_in[6],  *ln2a_b = (const float*)d_in[7];
  const float* ln2b_g = (const float*)d_in[8],  *ln2b_b = (const float*)d_in[9];
  const float* sa_qw = (const float*)d_in[10], *sa_qb = (const float*)d_in[11];
  const float* sa_kw = (const float*)d_in[12], *sa_kb = (const float*)d_in[13];
  const float* sa_vw = (const float*)d_in[14], *sa_vb = (const float*)d_in[15];
  const float* sa_pw = (const float*)d_in[16], *sa_pb = (const float*)d_in[17];
  const float* ca_qw = (const float*)d_in[18], *ca_qb = (const float*)d_in[19];
  const float* ca_kw = (const float*)d_in[20], *ca_kb = (const float*)d_in[21];
  const float* ca_vw = (const float*)d_in[22], *ca_vb = (const float*)d_in[23];
  const float* ca_pw = (const float*)d_in[24], *ca_pb = (const float*)d_in[25];
  const float* m1_w1 = (const float*)d_in[26], *m1_b1 = (const float*)d_in[27];
  const float* m1_w2 = (const float*)d_in[28], *m1_b2 = (const float*)d_in[29];
  const float* m2_w1 = (const float*)d_in[30], *m2_b1 = (const float*)d_in[31];
  const float* m2_w2 = (const float*)d_in[32], *m2_b2 = (const float*)d_in[33];
  float* out = (float*)d_out;

  const int S = S_LEN, C = C_DIM, M = M_DIM;
  size_t SC = (size_t)S * C;
  float* ws  = (float*)d_ws;
  float* xn  = ws;            // S*C f32
  float* qb  = ws + SC;       // [H][S][64] bf16 (overallocated as SC floats)
  float* kb  = ws + 2 * SC;   // [H][S][64] bf16
  float* vb  = ws + 3 * SC;   // [H][64][S] bf16 (transposed)
  float* ctx = ws + 4 * SC;   // S*C f32
  float* x1  = ws + 5 * SC;
  float* x2  = ws + 6 * SC;
  float* x3  = ws + 7 * SC;
  float* hid = ws + 8 * SC;   // S*M f32 = 4*SC

  dim3 blk(256);
  dim3 gCC(C / TN, S / TM);
  dim3 gCM(M / TN, S / TM);
  dim3 gAttn(S / 128, NHEAD);
  const float qscale = 0.125f;  // hd^-0.5 = 1/8

  // ---- self-attention block (causal) ----
  ln_kernel<<<S, blk, 0, stream>>>(feat, ln1a_g, ln1a_b, xn);
  gemm_kernel<3><<<gCC, blk, 0, stream>>>(xn, sa_qw, sa_qb, nullptr, qb, S, C, C, qscale);
  gemm_kernel<3><<<gCC, blk, 0, stream>>>(xn, sa_kw, sa_kb, nullptr, kb, S, C, C, 1.0f);
  gemm_kernel<4><<<gCC, blk, 0, stream>>>(xn, sa_vw, sa_vb, nullptr, vb, S, C, C, 1.0f);
  attn_kernel<1><<<gAttn, blk, 0, stream>>>((const __bf16*)qb, (const __bf16*)kb,
                                            (const __bf16*)vb, ctx);
  gemm_kernel<2><<<gCC, blk, 0, stream>>>(ctx, sa_pw, sa_pb, feat, x1, S, C, C, 1.0f);
  // ---- MLP 1 ----
  ln_kernel<<<S, blk, 0, stream>>>(x1, ln1b_g, ln1b_b, xn);
  gemm_kernel<1><<<gCM, blk, 0, stream>>>(xn, m1_w1, m1_b1, nullptr, hid, S, M, C, 1.0f);
  gemm_kernel<2><<<gCC, blk, 0, stream>>>(hid, m1_w2, m1_b2, x1, x2, S, C, M, 1.0f);
  // ---- cross-attention (k = v = encoder_output, no mask) ----
  ln_kernel<<<S, blk, 0, stream>>>(x2, ln2a_g, ln2a_b, xn);
  gemm_kernel<3><<<gCC, blk, 0, stream>>>(xn, ca_qw, ca_qb, nullptr, qb, S, C, C, qscale);
  gemm_kernel<3><<<gCC, blk, 0, stream>>>(enc, ca_kw, ca_kb, nullptr, kb, S, C, C, 1.0f);
  gemm_kernel<4><<<gCC, blk, 0, stream>>>(enc, ca_vw, ca_vb, nullptr, vb, S, C, C, 1.0f);
  attn_kernel<0><<<gAttn, blk, 0, stream>>>((const __bf16*)qb, (const __bf16*)kb,
                                            (const __bf16*)vb, ctx);
  gemm_kernel<2><<<gCC, blk, 0, stream>>>(ctx, ca_pw, ca_pb, x2, x3, S, C, C, 1.0f);
  // ---- MLP 2 ----
  ln_kernel<<<S, blk, 0, stream>>>(x3, ln2b_g, ln2b_b, xn);
  gemm_kernel<1><<<gCM, blk, 0, stream>>>(xn, m2_w1, m2_b1, nullptr, hid, S, M, C, 1.0f);
  gemm_kernel<2><<<gCC, blk, 0, stream>>>(hid, m2_w2, m2_b2, x3, out, S, C, M, 1.0f);

  (void)in_sizes; (void)n_in; (void)out_size; (void)ws_size;
}
